// WindowMCA_35201551958430
// MI455X (gfx1250) — matrix operand order repaired and checked by reading, not compiled, not run on hardware
//
#include <hip/hip_runtime.h>

// ---------- types ----------
typedef __attribute__((ext_vector_type(8)))  float          v8f;
typedef __attribute__((ext_vector_type(16))) __bf16         v16bf;
typedef __attribute__((ext_vector_type(8)))  unsigned short su8;

union Frag { v16bf b; su8 s[2]; unsigned short u[16]; };

#define NTOK  49
#define NP    64
#define CDIM  256
#define NH    8
#define HD    32
#define NWIN  64
#define SCALE 0.17677669529663687f   // 32^-0.5

#define XS  264   // padded row stride (halfs) for [64][256] LDS buffers
#define PS  72    // padded row stride (halfs) for [64][64] prob buffers
#define VTS 72    // padded row stride (halfs) for vT [8*32][64] buffer

__device__ __forceinline__ unsigned short f2bf(float f) {
    unsigned u = __float_as_uint(f);
    u += 0x7FFFu + ((u >> 16) & 1u);          // round-to-nearest-even
    return (unsigned short)(u >> 16);
}

// ---------- prep 1: pack weights into per-lane WMMA B-fragment order ----------
// wpk[t][kt][nt][lane][e] = W_t[n][k], n = nt*16 + (lane&15),
// k = kt*32 + (lane>>4)*8 + (e&7) + (e>>3)*16   (B 32x16: table K order per half-wave)
__global__ void __launch_bounds__(256) pack_w_kernel(
    const float* __restrict__ qw, const float* __restrict__ kw,
    const float* __restrict__ vw, const float* __restrict__ ow,
    unsigned short* __restrict__ wpk)
{
    int gid  = blockIdx.x * 256 + threadIdx.x;        // 262144 total
    int e    = gid & 15;
    int lane = (gid >> 4) & 31;
    int nt   = (gid >> 9) & 15;
    int kt   = (gid >> 13) & 7;
    int t    = gid >> 16;
    const float* W = (t == 0) ? qw : (t == 1) ? kw : (t == 2) ? vw : ow;
    int n = nt * 16 + (lane & 15);
    int k = kt * 32 + (lane >> 4) * 8 + (e & 7) + ((e >> 3) << 4);
    wpk[gid] = f2bf(W[n * CDIM + k]);
}

// ---------- prep 2: fused (rel-pos-bias + window-mask), packed in C-fragment order ----
// bmf[wi][h][mt][nt][lane][r]: element (m,n) of padded 64x64 score bias,
// m = mt*16 + r + 8*(lane>=16), n = nt*16 + (lane&15); padded cols -> -1e30
__global__ void __launch_bounds__(256) pack_bm_kernel(
    const float* __restrict__ mask, const float* __restrict__ rpb,
    const int* __restrict__ relidx, float* __restrict__ bmf)
{
    int gid  = blockIdx.x * 256 + threadIdx.x;        // 2097152 total
    int r    = gid & 7;
    int lane = (gid >> 3) & 31;
    int nt   = (gid >> 8) & 3;
    int mt   = (gid >> 10) & 3;
    int h    = (gid >> 12) & 7;
    int wi   = gid >> 15;
    int m = mt * 16 + r + ((lane >> 4) << 3);
    int n = nt * 16 + (lane & 15);
    float v;
    if (n >= NTOK)      v = -1e30f;
    else if (m >= NTOK) v = 0.0f;
    else                v = rpb[relidx[m * NTOK + n] * NH + h]
                          + mask[((size_t)wi * NTOK + m) * NTOK + n];
    bmf[gid] = v;
}

// ---------- fused window attention: one block per window, one wave per head ----------
__global__ void __launch_bounds__(256) wmca_fused_kernel(
    const float* __restrict__ xq, const float* __restrict__ xk, const float* __restrict__ xv,
    const unsigned short* __restrict__ wpk,
    const float* __restrict__ qb, const float* __restrict__ kb, const float* __restrict__ vb,
    const float* __restrict__ ob, const float* __restrict__ bmf,
    float* __restrict__ out)
{
    extern __shared__ unsigned short smem[];
    unsigned short* xin = smem;                             // [64][XS] staged input
    unsigned short* qs  = smem + 1 * NP * XS;               // q (scaled); later attn-out
    unsigned short* ks  = smem + 2 * NP * XS;               // k
    unsigned short* vs  = smem + 3 * NP * XS;               // v (row-major)
    unsigned short* pbf = smem + 4 * NP * XS;               // per-wave [64][PS] probs
    unsigned short* vtp = pbf + NH * NP * PS;               // vT [8*32][VTS]

    const int b    = blockIdx.x;
    const int tid  = threadIdx.x;
    const int wave = tid >> 5;
    const int lane = tid & 31;
    const int lo   = lane & 15;
    const int hi   = lane >> 4;

    // ================= Phase 1: Q/K/V projections (t fully unrolled) =================
#pragma unroll
    for (int t = 0; t < 3; ++t) {
        const float* xp = ((t == 0) ? xq : (t == 1) ? xk : xv) + (size_t)b * NTOK * CDIM;
        const float* bias = (t == 0) ? qb : (t == 1) ? kb : vb;
        unsigned short* dst = (t == 0) ? qs : (t == 1) ? ks : vs;

        // cooperative stage: fp32 -> bf16 (float4 loads, ushort4 stores), pad rows
#pragma unroll
        for (int j = 0; j < 16; ++j) {
            int lin = j * 256 + tid;          // float4 index over [64][64]
            int row = lin >> 6;
            int c4  = lin & 63;
            ushort4 o4 = {0, 0, 0, 0};
            if (row < NTOK) {
                float4 f4 = *(const float4*)(xp + row * CDIM + c4 * 4);
                o4.x = f2bf(f4.x); o4.y = f2bf(f4.y); o4.z = f2bf(f4.z); o4.w = f2bf(f4.w);
            }
            *(ushort4*)(xin + row * XS + c4 * 4) = o4;
        }
        // prefetch next projection's input while this one computes
        if (t < 2) {
            const float* xn = ((t == 0) ? xk : xv) + (size_t)b * NTOK * CDIM;
#pragma unroll
            for (int j = 0; j < 2; ++j)
                __builtin_prefetch(xn + (size_t)(j * 256 + tid) * 32, 0, 0);
        }
        __syncthreads();

        for (int nti = 0; nti < 2; ++nti) {
            const int nt  = wave * 2 + nti;               // 16 n-tiles over 8 waves
            const int col = nt * 16 + lo;
            const float bcol = bias[col];
            for (int mt = 0; mt < 4; ++mt) {
                v8f acc0 = {}, acc1 = {};                 // two chains: even/odd K-step
#pragma unroll
                for (int kt = 0; kt < 8; ++kt) {
                    Frag a, w;
                    const unsigned short* ap = xin + (mt * 16 + lo) * XS + kt * 32 + hi * 8;
                    a.s[0] = *(const su8*)ap;
                    a.s[1] = *(const su8*)(ap + 16);
                    const su8* wp = (const su8*)(wpk + ((((t * 8 + kt) * 16 + nt) * 32 + lane) << 4));
                    w.s[0] = wp[0];
                    w.s[1] = wp[1];
                    if (kt & 1)
                        acc1 = __builtin_amdgcn_wmma_f32_16x16x32_bf16(false, a.b, false, w.b,
                                                                       (short)0, acc1, false, false);
                    else
                        acc0 = __builtin_amdgcn_wmma_f32_16x16x32_bf16(false, a.b, false, w.b,
                                                                       (short)0, acc0, false, false);
                }
#pragma unroll
                for (int r = 0; r < 8; ++r) {
                    float f = acc0[r] + acc1[r] + bcol;
                    if (t == 0) f *= SCALE;               // statically folded into q copy
                    const int m = mt * 16 + r + hi * 8;
                    const unsigned short bfv = f2bf(f);
                    dst[m * XS + col] = bfv;
                    if (t == 2) {
                        // transposed copy: wave w owns head w's columns exactly
                        const int dl = nti * 16 + lo;     // d within head
                        vtp[(wave * HD + dl) * VTS + m] = bfv;
                    }
                }
            }
        }
        __syncthreads();
    }

    // ================= Phase 2: scores + softmax (wave == head) =================
    const int h  = wave;
    const int ch = h * HD;
    const float* bmp = bmf + ((((size_t)(b & (NWIN - 1))) * NH + h) << 12);  // 4096 floats
    unsigned short* pw = pbf + wave * NP * PS;

    Frag kf[4];
#pragma unroll
    for (int nt = 0; nt < 4; ++nt) {
        const unsigned short* kp = ks + (nt * 16 + lo) * XS + ch + hi * 8;
        kf[nt].s[0] = *(const su8*)kp;
        kf[nt].s[1] = *(const su8*)(kp + 16);
    }
    for (int mt = 0; mt < 4; ++mt) {
        Frag qa;
        const unsigned short* qp = qs + (mt * 16 + lo) * XS + ch + hi * 8;
        qa.s[0] = *(const su8*)qp;
        qa.s[1] = *(const su8*)(qp + 16);
        v8f sc[4];
#pragma unroll
        for (int nt = 0; nt < 4; ++nt) {
            v8f z = {};
            sc[nt] = __builtin_amdgcn_wmma_f32_16x16x32_bf16(false, qa.b, false, kf[nt].b,
                                                             (short)0, z, false, false);
        }
        // bias+mask: one 32-B fragment-ordered vector load per tile
#pragma unroll
        for (int nt = 0; nt < 4; ++nt) {
            const v8f bv = *(const v8f*)(bmp + (size_t)(((mt * 4 + nt) * 32 + lane) << 3));
            sc[nt] += bv;
        }
        // softmax: a row lives in one VGPR slot across 16 lanes of a half-wave
#pragma unroll
        for (int r = 0; r < 8; ++r) {
            float mx = fmaxf(fmaxf(sc[0][r], sc[1][r]), fmaxf(sc[2][r], sc[3][r]));
            for (int o = 1; o < 16; o <<= 1) mx = fmaxf(mx, __shfl_xor(mx, o, 32));
            float s = 0.0f;
#pragma unroll
            for (int nt = 0; nt < 4; ++nt) {
                float e = __expf(sc[nt][r] - mx);
                sc[nt][r] = e;
                s += e;
            }
            for (int o = 1; o < 16; o <<= 1) s += __shfl_xor(s, o, 32);
            const float inv = 1.0f / s;
            const int m = mt * 16 + r + hi * 8;
#pragma unroll
            for (int nt = 0; nt < 4; ++nt)
                pw[m * PS + nt * 16 + lo] = f2bf(sc[nt][r] * inv);
        }
    }
    __syncthreads();   // all waves done reading qs -> safe to alias as attn output

    // ================= Phase 3: attn @ V (B-frags from contiguous vT rows) =========
    unsigned short* xo = qs;   // alias
#pragma unroll
    for (int mt = 0; mt < 4; ++mt) {
#pragma unroll
        for (int dt = 0; dt < 2; ++dt) {
            v8f acc = {};
#pragma unroll
            for (int kx = 0; kx < 2; ++kx) {
                Frag a, bv;
                const unsigned short* pp = pw + (mt * 16 + lo) * PS + kx * 32 + hi * 8;
                a.s[0] = *(const su8*)pp;
                a.s[1] = *(const su8*)(pp + 16);
                const int dl = dt * 16 + lo;              // d within head
                const unsigned short* vp = vtp + (h * HD + dl) * VTS + kx * 32 + hi * 8;
                bv.s[0] = *(const su8*)vp;
                bv.s[1] = *(const su8*)(vp + 16);
                acc = __builtin_amdgcn_wmma_f32_16x16x32_bf16(false, a.b, false, bv.b,
                                                              (short)0, acc, false, false);
            }
            const int col = ch + dt * 16 + lo;
#pragma unroll
            for (int r = 0; r < 8; ++r)
                xo[(mt * 16 + r + hi * 8) * XS + col] = f2bf(acc[r]);
        }
    }
    __syncthreads();

    // ================= Phase 4: output projection, fp32 store =================
    float* op = out + (size_t)b * NTOK * CDIM;
    for (int nti = 0; nti < 2; ++nti) {
        const int nt  = wave * 2 + nti;
        const int col = nt * 16 + lo;
        const float bo = ob[col];
#pragma unroll
        for (int mt = 0; mt < 4; ++mt) {
            v8f acc0 = {}, acc1 = {};
#pragma unroll
            for (int kt = 0; kt < 8; ++kt) {
                Frag a, w;
                const unsigned short* ap = xo + (mt * 16 + lo) * XS + kt * 32 + hi * 8;
                a.s[0] = *(const su8*)ap;
                a.s[1] = *(const su8*)(ap + 16);
                const su8* wp = (const su8*)(wpk + ((((3 * 8 + kt) * 16 + nt) * 32 + lane) << 4));
                w.s[0] = wp[0];
                w.s[1] = wp[1];
                if (kt & 1)
                    acc1 = __builtin_amdgcn_wmma_f32_16x16x32_bf16(false, a.b, false, w.b,
                                                                   (short)0, acc1, false, false);
                else
                    acc0 = __builtin_amdgcn_wmma_f32_16x16x32_bf16(false, a.b, false, w.b,
                                                                   (short)0, acc0, false, false);
            }
#pragma unroll
            for (int r = 0; r < 8; ++r) {
                const int m = mt * 16 + r + hi * 8;      // guard folds away for mt < 3
                if (mt < 3 || m < NTOK) op[m * CDIM + col] = acc0[r] + acc1[r] + bo;
            }
        }
    }
}

// ---------- launch ----------
extern "C" void kernel_launch(void* const* d_in, const int* in_sizes, int n_in,
                              void* d_out, int out_size, void* d_ws, size_t ws_size,
                              hipStream_t stream)
{
    (void)in_sizes; (void)n_in; (void)out_size; (void)ws_size;
    const float* query = (const float*)d_in[0];
    const float* key_  = (const float*)d_in[1];
    const float* value = (const float*)d_in[2];
    const float* mask  = (const float*)d_in[3];
    const float* q_w   = (const float*)d_in[4];
    const float* q_b   = (const float*)d_in[5];
    const float* k_w   = (const float*)d_in[6];
    const float* k_b   = (const float*)d_in[7];
    const float* v_w   = (const float*)d_in[8];
    const float* v_b   = (const float*)d_in[9];
    const float* o_w   = (const float*)d_in[10];
    const float* o_b   = (const float*)d_in[11];
    const float* rpb   = (const float*)d_in[12];
    const int*   ridx  = (const int*)d_in[13];

    unsigned short* wpk = (unsigned short*)d_ws;                 // 512 KB packed weights
    float*          bmf = (float*)((char*)d_ws + 524288);        // 8 MB fused bias+mask

    pack_w_kernel <<<1024, 256, 0, stream>>>(q_w, k_w, v_w, o_w, wpk);
    pack_bm_kernel<<<8192, 256, 0, stream>>>(mask, rpb, ridx, bmf);

    const size_t smem = (size_t)(4 * NP * XS + NH * NP * PS + NH * HD * VTS)
                      * sizeof(unsigned short);                  // ~240 KB
    wmca_fused_kernel<<<4096, 256, smem, stream>>>(query, key_, value, wpk,
                                                   q_b, k_b, v_b, o_b, bmf, (float*)d_out);
}
